// GLCA_MixAttention_Serial_88819923681999
// MI455X (gfx1250) — matrix operand order repaired and checked by reading, not compiled, not run on hardware
//
#include <hip/hip_runtime.h>

// ---------------- CDNA5 WMMA types & helpers ----------------
typedef __bf16 bf16_t;
typedef bf16_t v16bf __attribute__((ext_vector_type(16)));
typedef float  v8f   __attribute__((ext_vector_type(8)));

union FragW { v16bf v; uint4 q[2]; };
union U4w  { uint4 q; unsigned short u[8]; };

#if defined(__has_builtin)
#if __has_builtin(__builtin_amdgcn_cvt_pk_bf16_f32)
#define HAVE_CVT_PK_BF16 1
#endif
#if __has_builtin(__builtin_amdgcn_global_load_async_to_lds_b128)
#define HAVE_ASYNC_LDS 1
#endif
#endif

// pack two f32 -> packed bf16x2 (low = a, high = b)
__device__ __forceinline__ unsigned int pack2bf(float a, float b) {
#ifdef HAVE_CVT_PK_BF16
    union { decltype(__builtin_amdgcn_cvt_pk_bf16_f32(0.f, 0.f)) v; unsigned int u; } x;
    x.v = __builtin_amdgcn_cvt_pk_bf16_f32(a, b);
    return x.u;
#else
    // round-half-up then splice the two high halves with one v_perm_b32
    unsigned int au = __float_as_uint(a) + 0x8000u;
    unsigned int bu = __float_as_uint(b) + 0x8000u;
    return __builtin_amdgcn_perm(bu, au, 0x07060302u);  // {b[3],b[2],a[3],a[2]}
#endif
}

__device__ __forceinline__ v8f vzero() {
    v8f z = {0.f, 0.f, 0.f, 0.f, 0.f, 0.f, 0.f, 0.f};
    return z;
}

// 16-byte global -> LDS copy; async (GLOBAL_LOAD_ASYNC_TO_LDS_B128 / ASYNCcnt) if available
__device__ __forceinline__ void cp16_g2l(void* dst_lds, const void* src_gbl) {
#ifdef HAVE_ASYNC_LDS
    typedef int v4i_vs __attribute__((vector_size(16)));
    typedef __attribute__((address_space(1))) v4i_vs* gp_t;
    typedef __attribute__((address_space(3))) v4i_vs* lp_t;
    gp_t g = (gp_t)(unsigned long long)(size_t)src_gbl;
    lp_t l = (lp_t)(unsigned int)(size_t)dst_lds;
    __builtin_amdgcn_global_load_async_to_lds_b128(g, l, 0, 0);
#else
    *reinterpret_cast<uint4*>(dst_lds) = *reinterpret_cast<const uint4*>(src_gbl);
#endif
}
__device__ __forceinline__ void cp16_wait() {
#ifdef HAVE_ASYNC_LDS
#if __has_builtin(__builtin_amdgcn_s_wait_asynccnt)
    __builtin_amdgcn_s_wait_asynccnt(0);
#else
    asm volatile("s_wait_asynccnt 0x0" ::: "memory");
#endif
#endif
}

template <bool RA>
__device__ __forceinline__ v8f wmma_r(v16bf a, v16bf b, v8f c) {
    return __builtin_amdgcn_wmma_f32_16x16x32_bf16(
        false, a, false, b, (short)0, c, /*reuse_a=*/RA, /*reuse_b=*/false);
}

// A fragment (16x32, row-major p[row][k], pitch ldu elements, ldu % 8 == 0)
__device__ __forceinline__ v16bf frag_a_w(const unsigned short* p, int ldu, int lane) {
    const int r = lane & 15, sel = (lane >> 4) & 1;
    const unsigned short* base = p + r * ldu + sel * 8;
    FragW f;
    f.q[0] = *reinterpret_cast<const uint4*>(base);
    f.q[1] = *reinterpret_cast<const uint4*>(base + 16);
    return f.v;
}
// B fragment (32x16) from n-major source p[n][k] (pitch ldu, ldu % 8 == 0)
__device__ __forceinline__ v16bf frag_b_nk_w(const unsigned short* p, int ldu, int lane) {
    const int n = lane & 15, sel = (lane >> 4) & 1;
    const unsigned short* base = p + n * ldu + sel * 8;
    FragW f;
    f.q[0] = *reinterpret_cast<const uint4*>(base);
    f.q[1] = *reinterpret_cast<const uint4*>(base + 16);
    return f.v;
}

#define WMMA4(S, A, B0, B1, B2, B3)        \
    S[0] = wmma_r<false>(A, B0, S[0]);     \
    S[1] = wmma_r<true >(A, B1, S[1]);     \
    S[2] = wmma_r<true >(A, B2, S[2]);     \
    S[3] = wmma_r<true >(A, B3, S[3]);

#define WMMA4_Z(S, A, B0, B1, B2, B3)          \
    S[0] = wmma_r<false>(A, B0, vzero());      \
    S[1] = wmma_r<true >(A, B1, vzero());      \
    S[2] = wmma_r<true >(A, B2, vzero());      \
    S[3] = wmma_r<true >(A, B3, vzero());

// ---------------- Kernel 1: Y = X(f32) @ W(f32) -> head-major bf16 ----------------
__global__ __launch_bounds__(128)
void gemm_x_w_headmajor(const float* __restrict__ X, const float* __restrict__ W,
                        int ldw, unsigned short* __restrict__ Out,
                        int Nrows, int Kdim) {
    __shared__ __align__(16) unsigned short As[64][40];  // [row][k]
    __shared__ __align__(16) unsigned short Bt[64][40];  // transposed W tile [c][k]
    const int tid = threadIdx.x, lane = tid & 31, wave = tid >> 5;
    const int rowBase = blockIdx.y * 64, colBase = blockIdx.x * 64;

    v8f acc[4];
#pragma unroll
    for (int j = 0; j < 4; ++j) acc[j] = vzero();

    for (int kb = 0; kb < Kdim; kb += 32) {
        __syncthreads();
#pragma unroll
        for (int it = 0; it < 4; ++it) {            // A: 64x32 f32 -> bf16
            int u = tid + 128 * it;
            int r = u >> 3, k4 = (u & 7) * 4;
            float4 xv = *reinterpret_cast<const float4*>(
                X + (size_t)(rowBase + r) * Kdim + kb + k4);
            *reinterpret_cast<uint2*>(&As[r][k4]) =
                make_uint2(pack2bf(xv.x, xv.y), pack2bf(xv.z, xv.w));
        }
#pragma unroll
        for (int it = 0; it < 4; ++it) {            // W: 32x64 f32 -> transposed bf16
            int u = tid + 128 * it;
            int k = u >> 4, c4 = (u & 15) * 4;
            float4 wv = *reinterpret_cast<const float4*>(
                W + (size_t)(kb + k) * ldw + colBase + c4);
            unsigned int p01 = pack2bf(wv.x, wv.y), p23 = pack2bf(wv.z, wv.w);
            Bt[c4 + 0][k] = (unsigned short)p01;
            Bt[c4 + 1][k] = (unsigned short)(p01 >> 16);
            Bt[c4 + 2][k] = (unsigned short)p23;
            Bt[c4 + 3][k] = (unsigned short)(p23 >> 16);
        }
        __syncthreads();
        v16bf a  = frag_a_w(&As[wave * 16][0], 40, lane);
        v16bf b0 = frag_b_nk_w(&Bt[0][0],  40, lane);
        v16bf b1 = frag_b_nk_w(&Bt[16][0], 40, lane);
        v16bf b2 = frag_b_nk_w(&Bt[32][0], 40, lane);
        v16bf b3 = frag_b_nk_w(&Bt[48][0], 40, lane);
        WMMA4(acc, a, b0, b1, b2, b3)
    }
    const int sel = lane >> 4, cn = lane & 15;
#pragma unroll
    for (int j = 0; j < 4; ++j)
#pragma unroll
        for (int r = 0; r < 8; r += 2) {
            int row = rowBase + wave * 16 + 8 * sel + r;
            int c = colBase + j * 16 + cn;
            int h = c >> 6, d = c & 63;
            unsigned int pk = pack2bf(acc[j][r], acc[j][r + 1]);
            Out[((size_t)h * Nrows + row) * 64 + d]     = (unsigned short)pk;
            Out[((size_t)h * Nrows + row + 1) * 64 + d] = (unsigned short)(pk >> 16);
        }
}

// ---------------- Kernel 2: flash attention (global, per head) ----------------
__global__ __launch_bounds__(128)
void flash_attn_global(const unsigned short* __restrict__ Qh,
                       const unsigned short* __restrict__ Kh,
                       const unsigned short* __restrict__ Vh,
                       unsigned short* __restrict__ Xg,
                       int N, float scale) {
    __shared__ __align__(16) unsigned short Qs[64][72];  // [q][d]
    __shared__ __align__(16) unsigned short Ks[64][72];  // [j][d]
    __shared__ __align__(16) unsigned short Vt[64][72];  // transposed V tile [d][j]
    __shared__ __align__(16) unsigned short Pb[64][72];  // [q][j]

    const int tid = threadIdx.x, lane = tid & 31, wave = tid >> 5;
    const int sel = lane >> 4, cn = lane & 15;
    const int h = blockIdx.y, qBase = blockIdx.x * 64;
    const unsigned short* Qp = Qh + ((size_t)h * N + qBase) * 64;

#pragma unroll
    for (int it = 0; it < 4; ++it) {                // Q tile: async straight copy
        int u = tid + 128 * it;
        int r = u >> 3, d8 = (u & 7) * 8;
        cp16_g2l(&Qs[r][d8], Qp + r * 64 + d8);
    }

    v8f o[4];
    float m[8], l[8];
#pragma unroll
    for (int j = 0; j < 4; ++j) o[j] = vzero();
#pragma unroll
    for (int r = 0; r < 8; ++r) { m[r] = -1e30f; l[r] = 0.f; }

    for (int kt = 0; kt < N; kt += 64) {
        __syncthreads();
        const unsigned short* Kp = Kh + ((size_t)h * N + kt) * 64;
        const unsigned short* Vp = Vh + ((size_t)h * N + kt) * 64;
#pragma unroll
        for (int it = 0; it < 4; ++it) {
            int u = tid + 128 * it;
            int r = u >> 3, d8 = (u & 7) * 8;
            cp16_g2l(&Ks[r][d8], Kp + r * 64 + d8);           // K: async copy
            U4w vv; vv.q = *reinterpret_cast<const uint4*>(Vp + r * 64 + d8);
#pragma unroll
            for (int e = 0; e < 8; ++e) Vt[d8 + e][r] = vv.u[e];  // V: transpose
        }
        if (kt + 64 < N) {                          // global_prefetch_b8
            __builtin_prefetch(Kp + 4096 + tid * 32, 0, 1);
            __builtin_prefetch(Vp + 4096 + tid * 32, 0, 1);
        }
        cp16_wait();                                // s_wait_asynccnt 0
        __syncthreads();

        // S = Q * K^T  (first kh-step folds C to inline 0)
        v8f s[4];
        {
            v16bf a  = frag_a_w(&Qs[wave * 16][0], 72, lane);
            v16bf b0 = frag_b_nk_w(&Ks[0][0],  72, lane);
            v16bf b1 = frag_b_nk_w(&Ks[16][0], 72, lane);
            v16bf b2 = frag_b_nk_w(&Ks[32][0], 72, lane);
            v16bf b3 = frag_b_nk_w(&Ks[48][0], 72, lane);
            WMMA4_Z(s, a, b0, b1, b2, b3)
        }
        {
            v16bf a  = frag_a_w(&Qs[wave * 16][32], 72, lane);
            v16bf b0 = frag_b_nk_w(&Ks[0][32],  72, lane);
            v16bf b1 = frag_b_nk_w(&Ks[16][32], 72, lane);
            v16bf b2 = frag_b_nk_w(&Ks[32][32], 72, lane);
            v16bf b3 = frag_b_nk_w(&Ks[48][32], 72, lane);
            WMMA4(s, a, b0, b1, b2, b3)
        }

        // register online softmax: lane owns rows (wave*16 + 8*sel + r), r = 0..7;
        // a row's 64 cols live across the 16 lanes of this half -> xor masks 1,2,4,8
#pragma unroll
        for (int r = 0; r < 8; ++r) {
            float mx = fmaxf(fmaxf(s[0][r], s[1][r]), fmaxf(s[2][r], s[3][r]));
#pragma unroll
            for (int off = 1; off < 16; off <<= 1)
                mx = fmaxf(mx, __shfl_xor(mx, off, 32));
            mx *= scale;
            float mnew = fmaxf(m[r], mx);
            float corr = __expf(m[r] - mnew);
            int row = wave * 16 + 8 * sel + r;
            float p0 = __expf(s[0][r] * scale - mnew);
            float p1 = __expf(s[1][r] * scale - mnew);
            float p2 = __expf(s[2][r] * scale - mnew);
            float p3 = __expf(s[3][r] * scale - mnew);
            unsigned int pk01 = pack2bf(p0, p1), pk23 = pack2bf(p2, p3);
            Pb[row][cn]      = (unsigned short)pk01;
            Pb[row][16 + cn] = (unsigned short)(pk01 >> 16);
            Pb[row][32 + cn] = (unsigned short)pk23;
            Pb[row][48 + cn] = (unsigned short)(pk23 >> 16);
            float psum = (p0 + p1) + (p2 + p3);
#pragma unroll
            for (int off = 1; off < 16; off <<= 1)
                psum += __shfl_xor(psum, off, 32);
#pragma unroll
            for (int j = 0; j < 4; ++j) o[j][r] *= corr;
            l[r] = l[r] * corr + psum;
            m[r] = mnew;
        }

        // O += P @ V   (B element (k=j, n=d) = Vt[d][j] -> n-major pattern)
#pragma unroll
        for (int kh = 0; kh < 64; kh += 32) {
            v16bf a  = frag_a_w(&Pb[wave * 16][kh], 72, lane);
            v16bf b0 = frag_b_nk_w(&Vt[0][kh],  72, lane);
            v16bf b1 = frag_b_nk_w(&Vt[16][kh], 72, lane);
            v16bf b2 = frag_b_nk_w(&Vt[32][kh], 72, lane);
            v16bf b3 = frag_b_nk_w(&Vt[48][kh], 72, lane);
            WMMA4(o, a, b0, b1, b2, b3)
        }
    }

    unsigned short* Op = Xg + ((size_t)h * N + qBase) * 64;
#pragma unroll
    for (int r = 0; r < 8; r += 2) {
        float inv0 = 1.f / l[r], inv1 = 1.f / l[r + 1];
        int rr = wave * 16 + 8 * sel + r;
#pragma unroll
        for (int j = 0; j < 4; ++j) {
            unsigned int pk = pack2bf(o[j][r] * inv0, o[j][r + 1] * inv1);
            Op[(size_t)rr * 64 + j * 16 + cn]       = (unsigned short)pk;
            Op[(size_t)(rr + 1) * 64 + j * 16 + cn] = (unsigned short)(pk >> 16);
        }
    }
}

// ---------------- Kernel 3: local 64-token block attention + permute ----------------
// block b = h*64 + w over head-major Xg; out[n = d*64 + i, c = h*64 + w] = O[i][d]
__global__ __launch_bounds__(128)
void local_block_attn(const unsigned short* __restrict__ Xg,
                      unsigned short* __restrict__ XlNC,
                      int N, float scale) {
    __shared__ __align__(16) unsigned short Xs[64][72];  // [i][d]
    __shared__ __align__(16) unsigned short Xt[64][72];  // transposed [d][i]
    __shared__ __align__(16) unsigned short Pb[64][72];

    const int tid = threadIdx.x, lane = tid & 31, wave = tid >> 5;
    const int sel = lane >> 4, cn = lane & 15;
    const int blk = blockIdx.x, h = blk >> 6, w = blk & 63;
    const unsigned short* Xp = Xg + ((size_t)h * N + w * 64) * 64;

#pragma unroll
    for (int it = 0; it < 4; ++it) {
        int u = tid + 128 * it;
        int r = u >> 3, d8 = (u & 7) * 8;
        cp16_g2l(&Xs[r][d8], Xp + r * 64 + d8);     // natural: async copy
        U4w vv; vv.q = *reinterpret_cast<const uint4*>(Xp + r * 64 + d8);
#pragma unroll
        for (int e = 0; e < 8; ++e) Xt[d8 + e][r] = vv.u[e];
    }
    cp16_wait();
    __syncthreads();

    v8f s[4];
    {
        v16bf a  = frag_a_w(&Xs[wave * 16][0], 72, lane);
        v16bf b0 = frag_b_nk_w(&Xs[0][0],  72, lane);
        v16bf b1 = frag_b_nk_w(&Xs[16][0], 72, lane);
        v16bf b2 = frag_b_nk_w(&Xs[32][0], 72, lane);
        v16bf b3 = frag_b_nk_w(&Xs[48][0], 72, lane);
        WMMA4_Z(s, a, b0, b1, b2, b3)
    }
    {
        v16bf a  = frag_a_w(&Xs[wave * 16][32], 72, lane);
        v16bf b0 = frag_b_nk_w(&Xs[0][32],  72, lane);
        v16bf b1 = frag_b_nk_w(&Xs[16][32], 72, lane);
        v16bf b2 = frag_b_nk_w(&Xs[32][32], 72, lane);
        v16bf b3 = frag_b_nk_w(&Xs[48][32], 72, lane);
        WMMA4(s, a, b0, b1, b2, b3)
    }

    float linv[8];
#pragma unroll
    for (int r = 0; r < 8; ++r) {
        float mx = fmaxf(fmaxf(s[0][r], s[1][r]), fmaxf(s[2][r], s[3][r]));
#pragma unroll
        for (int off = 1; off < 16; off <<= 1)
            mx = fmaxf(mx, __shfl_xor(mx, off, 32));
        mx *= scale;
        int row = wave * 16 + 8 * sel + r;
        float p0 = __expf(s[0][r] * scale - mx);
        float p1 = __expf(s[1][r] * scale - mx);
        float p2 = __expf(s[2][r] * scale - mx);
        float p3 = __expf(s[3][r] * scale - mx);
        unsigned int pk01 = pack2bf(p0, p1), pk23 = pack2bf(p2, p3);
        Pb[row][cn]      = (unsigned short)pk01;
        Pb[row][16 + cn] = (unsigned short)(pk01 >> 16);
        Pb[row][32 + cn] = (unsigned short)pk23;
        Pb[row][48 + cn] = (unsigned short)(pk23 >> 16);
        float psum = (p0 + p1) + (p2 + p3);
#pragma unroll
        for (int off = 1; off < 16; off <<= 1)
            psum += __shfl_xor(psum, off, 32);
        linv[r] = 1.f / psum;
    }

    v8f o[4];
    {
        v16bf a  = frag_a_w(&Pb[wave * 16][0], 72, lane);
        v16bf b0 = frag_b_nk_w(&Xt[0][0],  72, lane);
        v16bf b1 = frag_b_nk_w(&Xt[16][0], 72, lane);
        v16bf b2 = frag_b_nk_w(&Xt[32][0], 72, lane);
        v16bf b3 = frag_b_nk_w(&Xt[48][0], 72, lane);
        WMMA4_Z(o, a, b0, b1, b2, b3)
    }
    {
        v16bf a  = frag_a_w(&Pb[wave * 16][32], 72, lane);
        v16bf b0 = frag_b_nk_w(&Xt[0][32],  72, lane);
        v16bf b1 = frag_b_nk_w(&Xt[16][32], 72, lane);
        v16bf b2 = frag_b_nk_w(&Xt[32][32], 72, lane);
        v16bf b3 = frag_b_nk_w(&Xt[48][32], 72, lane);
        WMMA4(o, a, b0, b1, b2, b3)
    }
    // scatter with the source's faithful permute
#pragma unroll
    for (int r = 0; r < 8; r += 2) {
        int i = wave * 16 + 8 * sel + r;
#pragma unroll
        for (int j = 0; j < 4; ++j) {
            int d = j * 16 + cn;
            unsigned int pk = pack2bf(o[j][r] * linv[r], o[j][r + 1] * linv[r + 1]);
            XlNC[(size_t)(d * 64 + i) * 768 + h * 64 + w]     = (unsigned short)pk;
            XlNC[(size_t)(d * 64 + i + 1) * 768 + h * 64 + w] = (unsigned short)(pk >> 16);
        }
    }
}

// ---------------- Kernel 4: Y = Xl(bf16) @ Wproj(f32) + b -> f32 out ----------------
__global__ __launch_bounds__(128)
void gemm_proj(const unsigned short* __restrict__ A,
               const float* __restrict__ W, const float* __restrict__ bias,
               float* __restrict__ Out, int Nrows, int Kdim) {
    __shared__ __align__(16) unsigned short As[64][40];
    __shared__ __align__(16) unsigned short Bt[64][40];
    const int tid = threadIdx.x, lane = tid & 31, wave = tid >> 5;
    const int rowBase = blockIdx.y * 64, colBase = blockIdx.x * 64;

    v8f acc[4];
#pragma unroll
    for (int j = 0; j < 4; ++j) acc[j] = vzero();

    for (int kb = 0; kb < Kdim; kb += 32) {
        __syncthreads();
#pragma unroll
        for (int it = 0; it < 2; ++it) {            // A tile: bf16 async copy
            int u = tid + 128 * it;
            int r = u >> 2, k8 = (u & 3) * 8;
            cp16_g2l(&As[r][k8], A + (size_t)(rowBase + r) * Kdim + kb + k8);
        }
#pragma unroll
        for (int it = 0; it < 4; ++it) {            // W tile: transposed bf16
            int u = tid + 128 * it;
            int k = u >> 4, c4 = (u & 15) * 4;
            float4 wv = *reinterpret_cast<const float4*>(
                W + (size_t)(kb + k) * Kdim + colBase + c4);
            unsigned int p01 = pack2bf(wv.x, wv.y), p23 = pack2bf(wv.z, wv.w);
            Bt[c4 + 0][k] = (unsigned short)p01;
            Bt[c4 + 1][k] = (unsigned short)(p01 >> 16);
            Bt[c4 + 2][k] = (unsigned short)p23;
            Bt[c4 + 3][k] = (unsigned short)(p23 >> 16);
        }
        cp16_wait();
        __syncthreads();
        v16bf a  = frag_a_w(&As[wave * 16][0], 40, lane);
        v16bf b0 = frag_b_nk_w(&Bt[0][0],  40, lane);
        v16bf b1 = frag_b_nk_w(&Bt[16][0], 40, lane);
        v16bf b2 = frag_b_nk_w(&Bt[32][0], 40, lane);
        v16bf b3 = frag_b_nk_w(&Bt[48][0], 40, lane);
        WMMA4(acc, a, b0, b1, b2, b3)
    }
    const int sel = lane >> 4, cn = lane & 15;
#pragma unroll
    for (int j = 0; j < 4; ++j) {
        int c = colBase + j * 16 + cn;
        float bv = bias[c];
#pragma unroll
        for (int r = 0; r < 8; ++r) {
            int row = rowBase + wave * 16 + 8 * sel + r;
            Out[(size_t)row * Kdim + c] = acc[j][r] + bv;
        }
    }
}

// ---------------- Host launch ----------------
extern "C" void kernel_launch(void* const* d_in, const int* in_sizes, int n_in,
                              void* d_out, int out_size, void* d_ws, size_t ws_size,
                              hipStream_t stream) {
    (void)in_sizes; (void)n_in; (void)out_size; (void)ws_size;
    const float* x     = (const float*)d_in[0];
    const float* Wq    = (const float*)d_in[1];
    const float* Wkv   = (const float*)d_in[2];
    const float* Wproj = (const float*)d_in[3];
    const float* bproj = (const float*)d_in[4];

    const int N = 4096, C = 768, nh = 12, hd = 64;
    const float scale = 0.125f;                      // hd^-0.5

    unsigned short* ws = (unsigned short*)d_ws;
    const size_t per = (size_t)nh * N * hd;          // 6 MB bf16 each
    unsigned short* Qh = ws;
    unsigned short* Kh = Qh + per;
    unsigned short* Vh = Kh + per;
    unsigned short* Xg = Vh + per;
    unsigned short* Xl = Xg + per;                   // (N, C) bf16

    dim3 blk(128);
    dim3 gGemm(C / 64, N / 64);                      // (12, 64)

    gemm_x_w_headmajor<<<gGemm, blk, 0, stream>>>(x, Wq,      C,     Qh, N, C);
    gemm_x_w_headmajor<<<gGemm, blk, 0, stream>>>(x, Wkv,     2 * C, Kh, N, C);
    gemm_x_w_headmajor<<<gGemm, blk, 0, stream>>>(x, Wkv + C, 2 * C, Vh, N, C);

    flash_attn_global<<<dim3(N / 64, nh), blk, 0, stream>>>(Qh, Kh, Vh, Xg, N, scale);
    local_block_attn<<<dim3(nh * N / 64), blk, 0, stream>>>(Xg, Xl, N, scale);

    gemm_proj<<<gGemm, blk, 0, stream>>>(Xl, Wproj, bproj, (float*)d_out, N, C);
}
